// NeuralCDE_16217796509886
// MI455X (gfx1250) — matrix operand order repaired and checked by reading, not compiled, not run on hardware
//
#include <hip/hip_runtime.h>

#define BATCH 512
#define LSEQ  64
#define CT    32
#define CCH   33      // CT + time channel
#define SDIM  16
#define HDIM  192
#define ODIM  10
#define STEPS 63      // (L-1) * N_SUB, N_SUB = 1

#define BB        16          // batch rows per workgroup
#define NT1       12          // HDIM/16 N-tiles, layer 1
#define NT2       396         // (CCH*HDIM)/16 N-tiles, layer 2
#define KT        6           // HDIM/32 K-tiles
#define NWAVES    8
#define TPB       256

typedef __bf16 bf16_t;
typedef bf16_t v16bf __attribute__((ext_vector_type(16)));
typedef float  v8f   __attribute__((ext_vector_type(8)));

__device__ __forceinline__ unsigned short f2bf(float x) {
  union { float f; unsigned int i; } u; u.f = x;
  unsigned int r = u.i + 0x7FFFu + ((u.i >> 16) & 1u);   // round-to-nearest-even
  return (unsigned short)(r >> 16);
}

// A-fragment (16x32 bf16) from LDS row, contiguous layout:
// lane<16: K = k0..k0+15 ; lane>=16: k0 += 16
__device__ __forceinline__ v16bf frag_lds(const unsigned short* row, int k0) {
  union { v16bf v; uint4 q[2]; } f;
  f.q[0] = *(const uint4*)(row + k0);
  f.q[1] = *(const uint4*)(row + k0 + 8);
  return f.v;
}

// B-fragment (32x16 bf16) from pre-packed global: 32 contiguous bytes per lane
__device__ __forceinline__ v16bf frag_gbl(const unsigned short* p) {
  union { v16bf v; uint4 q[2]; } f;
  f.q[0] = ((const uint4*)p)[0];
  f.q[1] = ((const uint4*)p)[1];
  return f.v;
}

// ---------------------------------------------------------------------------
// Prep: round f_W1 / f_W2 to bf16 and pack fragment-major.
// Packed tile (nt*KT+kt): 512 bf16; entry [lane*16 + e] = W[n, k],
//   n = nt*16 + (lane&15), k = kt*32 + ((lane>=16)?16:0) + e   (B-matrix layout)
// ---------------------------------------------------------------------------
__global__ __launch_bounds__(TPB)
void cde_pack_weights(const float* __restrict__ fW1, const float* __restrict__ fW2,
                      unsigned short* __restrict__ pW1, unsigned short* __restrict__ pW2) {
  int wave = (blockIdx.x * TPB + threadIdx.x) >> 5;
  int lane = threadIdx.x & 31;
  const int totalTiles = NT1 * KT + NT2 * KT;   // 72 + 2376
  if (wave >= totalTiles) return;
  int n16   = lane & 15;
  int khalf = (lane >> 4) << 4;                 // 0 or 16
  if (wave < NT1 * KT) {
    int nt = wave / KT, kt = wave % KT;
    unsigned short* dst = pW1 + ((size_t)wave << 9) + (lane << 4);
    int n = nt * 16 + n16;
#pragma unroll
    for (int e = 0; e < 16; ++e) {
      int k = kt * 32 + khalf + e;
      dst[e] = f2bf(fW1[n * HDIM + k]);
    }
  } else {
    int t = wave - NT1 * KT;
    int nt = t / KT, kt = t % KT;
    unsigned short* dst = pW2 + ((size_t)t << 9) + (lane << 4);
    int n = nt * 16 + n16;
#pragma unroll
    for (int e = 0; e < 16; ++e) {
      int k = kt * 32 + khalf + e;
      dst[e] = f2bf(fW2[(size_t)n * HDIM + k]);
    }
  }
}

// ---------------------------------------------------------------------------
// Main: one workgroup integrates 16 batch rows through all 63 RK4 steps.
// ---------------------------------------------------------------------------
__global__ __launch_bounds__(TPB)
void cde_integrate(const float* __restrict__ static_data,
                   const float* __restrict__ temporal,
                   const float* __restrict__ obst,
                   const unsigned short* __restrict__ pW1,
                   const unsigned short* __restrict__ pW2,
                   const float* __restrict__ fb1, const float* __restrict__ fb2,
                   const float* __restrict__ tiW1, const float* __restrict__ tib1,
                   const float* __restrict__ tiW2, const float* __restrict__ tib2,
                   const float* __restrict__ stW1, const float* __restrict__ stb1,
                   const float* __restrict__ stW2, const float* __restrict__ stb2,
                   const float* __restrict__ oW1,  const float* __restrict__ ob1,
                   const float* __restrict__ oW2,  const float* __restrict__ ob2,
                   float* __restrict__ out) {
  __shared__ __align__(16) float zcur[BB * HDIM];            // z state, f32
  __shared__ __align__(16) float gacc[BB * HDIM];            // k_i accumulator (atomics)
  __shared__ __align__(16) float kacc[BB * HDIM];            // RK4 weighted sum
  __shared__ __align__(16) unsigned short zeb[BB * HDIM];    // bf16 z_eval (WMMA A)
  __shared__ __align__(16) unsigned short h1b[BB * HDIM];    // bf16 hidden (WMMA A)
  __shared__ float dxbuf[3][BB * CCH];                       // dxdt at ss=0, 0.5, 1
  __shared__ float x0s[BB * CCH];

  const int tid   = threadIdx.x;
  const int wave  = tid >> 5;
  const int lane  = tid & 31;
  const int m16   = lane & 15;            // row (A) / col (C,D) within tile
  const int half8 = (lane >> 4) << 3;     // C/D row offset: 0 or 8
  const int aoff  = (lane >> 4) << 4;     // A K-offset: 0 or 16
  const int bBase = blockIdx.x * BB;

  // ---- initial state: z0 = MLP(x(t0)) --------------------------------------
  for (int t = tid; t < BB * CCH; t += TPB) {
    int mm = t / CCH, c = t % CCH;
    int b = bBase + mm;
    x0s[t] = (c < CT) ? temporal[((size_t)b * LSEQ) * CT + c]
                      : obst[(size_t)b * LSEQ];
  }
  __syncthreads();
  for (int i = tid; i < BB * HDIM; i += TPB) {
    int mm = i / HDIM, n = i % HDIM;
    float a = tib1[n];
    for (int c = 0; c < CCH; ++c) a += tiW1[n * CCH + c] * x0s[mm * CCH + c];
    gacc[i] = a > 0.f ? a : 0.f;
  }
  __syncthreads();
  for (int i = tid; i < BB * HDIM; i += TPB) {
    int mm = i / HDIM, n = i % HDIM;
    float a = tib2[n];
    for (int k = 0; k < HDIM; ++k) a += tiW2[n * HDIM + k] * gacc[mm * HDIM + k];
    zcur[i] = a;
  }
  __syncthreads();

  const float cf[4] = {0.f, 0.5f, 0.5f, 1.f};
  const float wk[4] = {1.f, 2.f, 2.f, 1.f};

  // ---- RK4 scan ------------------------------------------------------------
  for (int step = 0; step < STEPS; ++step) {
    // Hermite-cubic dX/dt coefficients for this segment (dt = 1)
    for (int t = tid; t < BB * CCH; t += TPB) {
      int mm = t / CCH, c = t % CCH;
      int b = bBase + mm;
      float xa, xb, xp;
      if (c < CT) {
        const float* T = temporal + (size_t)b * LSEQ * CT + c;
        xa = T[(size_t)step * CT];
        xb = T[(size_t)(step + 1) * CT];
        xp = step ? T[(size_t)(step - 1) * CT] : 0.f;
      } else {
        const float* Ob = obst + (size_t)b * LSEQ;
        xa = Ob[step]; xb = Ob[step + 1]; xp = step ? Ob[step - 1] : 0.f;
      }
      float d  = xb - xa;
      float ma = step ? (xa - xp) : d;
      dxbuf[0][t] = ma;                       // ss = 0   -> m_a
      dxbuf[1][t] = 1.25f * d - 0.25f * ma;   // ss = 0.5
      dxbuf[2][t] = d;                        // ss = 1   -> m_b
    }
    for (int i = tid; i < BB * HDIM; i += TPB) kacc[i] = 0.f;
    __syncthreads();

    for (int ev = 0; ev < 4; ++ev) {
      // z_eval = z + c*k_prev -> bf16 ; zero k accumulator
      for (int i = tid; i < BB * HDIM; i += TPB) {
        float zv = zcur[i];
        if (ev > 0) zv += cf[ev] * gacc[i];
        zeb[i]  = f2bf(zv);
        gacc[i] = 0.f;
      }
      __syncthreads();

      // layer 1: h = relu(z_eval @ W1^T + b1)   [16 x 192]
      for (int nt = wave; nt < NT1; nt += NWAVES) {
        v8f acc = {};
#pragma unroll
        for (int kt = 0; kt < KT; ++kt) {
          v16bf a = frag_lds(&zeb[m16 * HDIM], kt * 32 + aoff);
          v16bf b = frag_gbl(pW1 + (((size_t)(nt * KT + kt)) << 9) + (lane << 4));
          acc = __builtin_amdgcn_wmma_f32_16x16x32_bf16(false, a, false, b,
                                                        (short)0, acc, false, false);
        }
        int n = nt * 16 + m16;
        float bias = fb1[n];
#pragma unroll
        for (int r = 0; r < 8; ++r) {
          float v = acc[r] + bias;
          h1b[(r + half8) * HDIM + n] = f2bf(v > 0.f ? v : 0.f);
        }
      }
      __syncthreads();

      // layer 2 fused with contraction, two independent WMMA chains per wave:
      //   k[b,h] += tanh((h @ W2^T + b2)[b, h*33+c]) * dx[b,c]
      const float* dx = dxbuf[(ev == 0) ? 0 : ((ev == 3) ? 2 : 1)];
      for (int nt0 = wave * 2; nt0 < NT2; nt0 += 2 * NWAVES) {
        const int nt1 = nt0 + 1;
        if (nt0 + 2 * NWAVES < NT2)
          __builtin_prefetch(pW2 + (((size_t)((nt0 + 2 * NWAVES) * KT)) << 9), 0, 0);
        v8f acc0 = {}, acc1 = {};
#pragma unroll
        for (int kt = 0; kt < KT; ++kt) {
          v16bf a  = frag_lds(&h1b[m16 * HDIM], kt * 32 + aoff);
          v16bf b0 = frag_gbl(pW2 + (((size_t)(nt0 * KT + kt)) << 9) + (lane << 4));
          v16bf b1 = frag_gbl(pW2 + (((size_t)(nt1 * KT + kt)) << 9) + (lane << 4));
          acc0 = __builtin_amdgcn_wmma_f32_16x16x32_bf16(false, a, false, b0,
                                                         (short)0, acc0, false, false);
          acc1 = __builtin_amdgcn_wmma_f32_16x16x32_bf16(false, a, false, b1,
                                                         (short)0, acc1, false, false);
        }
        {
          int j = nt0 * 16 + m16;
          float bias = fb2[j];
          int h = j / CCH;
          int c = j - h * CCH;
#pragma unroll
          for (int r = 0; r < 8; ++r) {
            int mr = r + half8;
            float v = tanhf(acc0[r] + bias);
            atomicAdd(&gacc[mr * HDIM + h], v * dx[mr * CCH + c]);
          }
        }
        {
          int j = nt1 * 16 + m16;
          float bias = fb2[j];
          int h = j / CCH;
          int c = j - h * CCH;
#pragma unroll
          for (int r = 0; r < 8; ++r) {
            int mr = r + half8;
            float v = tanhf(acc1[r] + bias);
            atomicAdd(&gacc[mr * HDIM + h], v * dx[mr * CCH + c]);
          }
        }
      }
      __syncthreads();

      // kacc += w_i * k_i ; finalize z after k4
      for (int i = tid; i < BB * HDIM; i += TPB) {
        kacc[i] += wk[ev] * gacc[i];
        if (ev == 3) zcur[i] += kacc[i] * (1.f / 6.f);
      }
      __syncthreads();
    }
  }

  // ---- readout: static embedding + output MLP ------------------------------
  for (int i = tid; i < BB * HDIM; i += TPB) {
    int mm = i / HDIM, n = i % HDIM;
    float a = stb1[n];
    for (int c = 0; c < SDIM; ++c)
      a += stW1[n * SDIM + c] * static_data[(size_t)(bBase + mm) * SDIM + c];
    gacc[i] = a > 0.f ? a : 0.f;
  }
  __syncthreads();
  for (int i = tid; i < BB * HDIM; i += TPB) {
    int mm = i / HDIM, n = i % HDIM;
    float a = stb2[n];
    for (int k = 0; k < HDIM; ++k) a += stW2[n * HDIM + k] * gacc[mm * HDIM + k];
    kacc[i] = a;                          // static embedding
  }
  __syncthreads();
  for (int i = tid; i < BB * HDIM; i += TPB) {
    int mm = i / HDIM, n = i % HDIM;
    float a = ob1[n];
    for (int k = 0; k < HDIM; ++k) {      // zcat = [zT, emb]
      a += oW1[n * 2 * HDIM + k] * zcur[mm * HDIM + k];
      a += oW1[n * 2 * HDIM + HDIM + k] * kacc[mm * HDIM + k];
    }
    gacc[i] = a > 0.f ? a : 0.f;
  }
  __syncthreads();
  for (int t = tid; t < BB * ODIM; t += TPB) {
    int mm = t / ODIM, o = t % ODIM;
    float a = ob2[o];
    for (int k = 0; k < HDIM; ++k) a += oW2[o * HDIM + k] * gacc[mm * HDIM + k];
    out[(size_t)(bBase + mm) * ODIM + o] = a;
  }
}

extern "C" void kernel_launch(void* const* d_in, const int* in_sizes, int n_in,
                              void* d_out, int out_size, void* d_ws, size_t ws_size,
                              hipStream_t stream) {
  (void)in_sizes; (void)n_in; (void)out_size; (void)ws_size;
  const float* static_data = (const float*)d_in[0];
  const float* temporal    = (const float*)d_in[1];
  const float* obst        = (const float*)d_in[2];
  const float* fW1  = (const float*)d_in[3];
  const float* fb1  = (const float*)d_in[4];
  const float* fW2  = (const float*)d_in[5];
  const float* fb2  = (const float*)d_in[6];
  const float* tiW1 = (const float*)d_in[7];
  const float* tib1 = (const float*)d_in[8];
  const float* tiW2 = (const float*)d_in[9];
  const float* tib2 = (const float*)d_in[10];
  const float* stW1 = (const float*)d_in[11];
  const float* stb1 = (const float*)d_in[12];
  const float* stW2 = (const float*)d_in[13];
  const float* stb2 = (const float*)d_in[14];
  const float* oW1  = (const float*)d_in[15];
  const float* ob1  = (const float*)d_in[16];
  const float* oW2  = (const float*)d_in[17];
  const float* ob2  = (const float*)d_in[18];

  unsigned short* pW1 = (unsigned short*)d_ws;             // 72 tiles * 1 KiB
  unsigned short* pW2 = pW1 + (size_t)NT1 * KT * 512;      // 2376 tiles * 1 KiB

  const int totalTiles = NT1 * KT + NT2 * KT;              // 2448 waves
  int packBlocks = (totalTiles + NWAVES - 1) / NWAVES;     // 306
  cde_pack_weights<<<packBlocks, TPB, 0, stream>>>(fW1, fW2, pW1, pW2);

  cde_integrate<<<BATCH / BB, TPB, 0, stream>>>(
      static_data, temporal, obst, pW1, pW2, fb1, fb2,
      tiW1, tib1, tiW2, tib2, stW1, stb1, stW2, stb2,
      oW1, ob1, oW2, ob2, (float*)d_out);
}
